// Local_Window_SA_18923625906510
// MI455X (gfx1250) — compile-verified, matrix operand order checked by reading, not run on hardware
//
#include <hip/hip_runtime.h>
#include <hip/hip_bf16.h>

typedef __attribute__((ext_vector_type(16))) __bf16 v16bf;
typedef __attribute__((ext_vector_type(8)))  float  v8f;

#define HEADS 8
#define DIM   256
#define HD    32
#define F3    768      // 3*DIM
#define MT    64       // tokens per workgroup
#define NWIN  16       // windows per workgroup (MT/4)

// hardware RNE f32 -> bf16
__device__ __forceinline__ unsigned short f2bf(float f) {
    union { __bf16 h; unsigned short s; } u;
    u.h = (__bf16)f;
    return u.s;
}
// pack two f32 -> one dword of 2 bf16 (v_cvt_pk_bf16_f32)
__device__ __forceinline__ unsigned pk2bf(float a, float b) {
    union { __bf16 h[2]; unsigned u; } p;
    p.h[0] = (__bf16)a;
    p.h[1] = (__bf16)b;
    return p.u;
}
// expand one dword (2 bf16) -> 2 floats with pure bit-ops
__device__ __forceinline__ void bfx2(unsigned u, float& lo, float& hi) {
    lo = __uint_as_float(u << 16);
    hi = __uint_as_float(u & 0xFFFF0000u);
}

// ---- tiny kernel: fp32 weights -> bf16 (stays resident in 192MB L2) ----
__global__ void convert_w_kernel(const float* __restrict__ w,
                                 unsigned short* __restrict__ wb) {
    int i = blockIdx.x * blockDim.x + threadIdx.x;   // 768*256 total
    wb[i] = f2bf(w[i]);
}

// ---- fused: window-partition + QKV GEMM (WMMA bf16) + attention + out ----
__global__ __launch_bounds__(256)
void swin_fused_kernel(const float* __restrict__ x,
                       const unsigned short* __restrict__ wb,   // [768][256] bf16
                       const float* __restrict__ bias_table,    // [9][8]
                       float* __restrict__ out) {
    // Single LDS block: xt FIRST (offset 0) so all A-fragment ds_load offsets
    // (< 32KB) fit the 16-bit DS immediate-offset field; qkvs at +32KB.
    __shared__ __align__(16) unsigned long long smem_raw[(32768 + 98304) / 8]; // 128 KB
    unsigned short* const xt   = (unsigned short*)smem_raw;           // [MT][DIM]
    unsigned short* const qkvs = (unsigned short*)smem_raw + 16384;   // [MT][F3]

    const int tid = threadIdx.x;

    // ================= Phase 1: load x tile (windowed order) ==============
    {
        int m  = tid >> 2;           // local token 0..63
        int k0 = (tid & 3) * 64;     // 64 features per thread
        int win_g = blockIdx.x * NWIN + (m >> 2);
        int n  = m & 3;              // position in 2x2 window
        int b  = win_g >> 10;
        int g  = win_g & 1023;
        int hh = ((g >> 5) << 1) + (n >> 1);
        int ww = ((g & 31) << 1) + (n & 1);
        const float* src = x + ((size_t)((b * 64 + hh) * 64 + ww)) * DIM + k0;
        unsigned* dst = (unsigned*)&xt[m * DIM + k0];
#pragma unroll
        for (int k = 0; k < 64; k += 4) {
            float4 f = *(const float4*)(src + k);
            dst[(k >> 1)]     = pk2bf(f.x, f.y);
            dst[(k >> 1) + 1] = pk2bf(f.z, f.w);
        }
    }
    __syncthreads();

    // ================= Phase 2: QKV GEMM via v_wmma_f32_16x16x32_bf16 =====
    // wave wv handles N-tiles {wv, wv+8, ..., wv+40}; all 4 M-tiles per N-tile
    const int wv  = tid >> 5;
    const int ln  = tid & 31;
    const int lhf = ln >> 4;      // lane half (K-half select)
    const int lm  = ln & 15;      // M (A) / N (B) index within tile

    // lane-fixed A base: all remaining terms (mt*8KB + k0*2 + {0,32}B) are
    // compile-time/loop constants that fold into DS immediate offsets
    const unsigned short* const abase = xt + lm * DIM + lhf * 8;

#pragma unroll 1
    for (int nti = 0; nti < 6; ++nti) {
        const int f0 = (wv + nti * 8) * 16;
        const unsigned short* const bbase = wb + (f0 + lm) * DIM + lhf * 8;
        v8f acc[4] = {};
#pragma unroll 2
        for (int kk = 0; kk < 8; ++kk) {
            const int k0 = kk * 32;
            // B fragment: 32x16 (KxN), lane lm = column; per-VGPR K pairs:
            // v<4 -> K=2v+8*lhf ; v>=4 -> K=16+2(v-4)+8*lhf (mirrors A layout)
            union { v16bf v; unsigned u[8]; } B;
#pragma unroll
            for (int v = 0; v < 4; ++v) {
                B.u[v]     = *(const unsigned*)(bbase + k0 + 2 * v);
                B.u[v + 4] = *(const unsigned*)(bbase + k0 + 16 + 2 * v);
            }
#pragma unroll
            for (int mt = 0; mt < 4; ++mt) {
                // A fragment: 16x32 (MxK), documented bf16 A layout
                union { v16bf v; unsigned u[8]; } A;
#pragma unroll
                for (int v = 0; v < 4; ++v) {
                    A.u[v]     = *(const unsigned*)(abase + mt * 16 * DIM + k0 + 2 * v);
                    A.u[v + 4] = *(const unsigned*)(abase + mt * 16 * DIM + k0 + 16 + 2 * v);
                }
                acc[mt] = __builtin_amdgcn_wmma_f32_16x16x32_bf16(
                    false, A.v, false, B.v, (short)0, acc[mt], false, false);
            }
        }
        // store D tiles to LDS qkv buffer (C/D layout: lane = N + M-half, vgpr r = M row)
        unsigned short* const qbase = qkvs + f0 + lm + (8 * lhf) * F3;
#pragma unroll
        for (int mt = 0; mt < 4; ++mt) {
#pragma unroll
            for (int r = 0; r < 8; r += 2) {
                unsigned p = pk2bf(acc[mt][r], acc[mt][r + 1]);
                qbase[(mt * 16 + r) * F3]     = (unsigned short)p;
                qbase[(mt * 16 + r + 1) * F3] = (unsigned short)(p >> 16);
            }
        }
    }
    __syncthreads();

    // ================= Phase 3: per-(window,head,row) attention ===========
    const float SCALE = 0.17677669529663687f;     // 32^-0.5
    // Swin WS=2 relative-position indices packed as nibbles (e = i*4+j)
    const unsigned long long RELPK = 0x4578346712450134ULL;
#pragma unroll 1
    for (int rep = 0; rep < 2; ++rep) {
        int item = tid + rep * 256;               // 0..511
        int win  = item >> 5;                     // 0..15
        int head = (item >> 2) & 7;
        int i    = item & 3;
        int mq   = win * 4 + i;

        float q[HD];
        {
            const unsigned* qr = (const unsigned*)&qkvs[mq * F3 + head * HD];
#pragma unroll
            for (int d = 0; d < HD; d += 2) bfx2(qr[d >> 1], q[d], q[d + 1]);
        }
        float s[4];
#pragma unroll
        for (int j = 0; j < 4; ++j) {
            const unsigned* kr = (const unsigned*)&qkvs[(win * 4 + j) * F3 + DIM + head * HD];
            float a = 0.f;
#pragma unroll
            for (int d = 0; d < HD; d += 2) {
                float k0, k1;
                bfx2(kr[d >> 1], k0, k1);
                a += q[d] * k0 + q[d + 1] * k1;
            }
            int ridx = (int)((RELPK >> ((i * 4 + j) * 4)) & 0xF);
            s[j] = a * SCALE + bias_table[ridx * HEADS + head];
        }
        float mx = fmaxf(fmaxf(s[0], s[1]), fmaxf(s[2], s[3]));
        float p[4], sum = 0.f;
#pragma unroll
        for (int j = 0; j < 4; ++j) { p[j] = __expf(s[j] - mx); sum += p[j]; }
        float inv = 1.f / sum;

        float o[HD];
#pragma unroll
        for (int d = 0; d < HD; ++d) o[d] = 0.f;
#pragma unroll
        for (int j = 0; j < 4; ++j) {
            const unsigned* vr = (const unsigned*)&qkvs[(win * 4 + j) * F3 + 2 * DIM + head * HD];
            float pj = p[j] * inv;
#pragma unroll
            for (int d = 0; d < HD; d += 2) {
                float v0, v1;
                bfx2(vr[d >> 1], v0, v1);
                o[d] += pj * v0;
                o[d + 1] += pj * v1;
            }
        }

        int win_g = blockIdx.x * NWIN + win;
        int b  = win_g >> 10;
        int g  = win_g & 1023;
        int hh = ((g >> 5) << 1) + (i >> 1);
        int ww = ((g & 31) << 1) + (i & 1);
        float* dst = out + ((size_t)((b * 64 + hh) * 64 + ww)) * DIM + head * HD;
#pragma unroll
        for (int d = 0; d < HD; d += 4)
            *(float4*)(dst + d) = make_float4(o[d], o[d + 1], o[d + 2], o[d + 3]);
    }
}

extern "C" void kernel_launch(void* const* d_in, const int* in_sizes, int n_in,
                              void* d_out, int out_size, void* d_ws, size_t ws_size,
                              hipStream_t stream) {
    const float* x          = (const float*)d_in[0];   // (64,64,64,256) f32
    const float* qkv_w      = (const float*)d_in[1];   // (768,256) f32
    const float* bias_table = (const float*)d_in[2];   // (9,8) f32
    float* out = (float*)d_out;

    unsigned short* wb = (unsigned short*)d_ws;        // 768*256 bf16 = 384 KB
    convert_w_kernel<<<F3, DIM, 0, stream>>>(qkv_w, wb);
    swin_fused_kernel<<<4096, 256, 0, stream>>>(x, wb, bias_table, out);
}
